// CrossModalBlock_8220567405006
// MI455X (gfx1250) — matrix operand order repaired and checked by reading, not compiled, not run on hardware
//
#include <hip/hip_runtime.h>
#include <hip/hip_bf16.h>

// ---------------------------------------------------------------- constants
constexpr int B  = 16;
constexpr int P  = 1024;
constexpr int Q  = 512;
constexpr int D  = 1024;
constexpr int H  = 16;
constexpr int DH = 64;          // D / H

typedef __attribute__((ext_vector_type(16))) __bf16 bf16x16;
typedef __attribute__((ext_vector_type(8)))  __bf16 bf16x8;
typedef __attribute__((ext_vector_type(8)))  float  f32x8;
typedef __attribute__((ext_vector_type(4)))  int    i32x4;

union Frag16 { bf16x16 v; bf16x8 h[2]; };

static __device__ inline f32x8 zero8() {
    f32x8 z = {0.f,0.f,0.f,0.f,0.f,0.f,0.f,0.f};
    return z;
}

static __device__ inline f32x8 wmma_bf16(bf16x16 a, bf16x16 b, f32x8 c) {
    // D = A(16x32) * B(32x16) + C, f32 accumulate
    return __builtin_amdgcn_wmma_f32_16x16x32_bf16(
        /*neg_a=*/false, a, /*neg_b=*/false, b,
        /*c_mod=*/(short)0, c, /*reuse_a=*/false, /*reuse_b=*/false);
}

// -------- CDNA5 async global->LDS path (guarded: falls back to sync staging)
#if defined(__has_builtin)
#  if __has_builtin(__builtin_amdgcn_global_load_async_to_lds_b128)
#    define HAVE_ASYNC_LDS 1
#  endif
#endif
#ifndef HAVE_ASYNC_LDS
#  define HAVE_ASYNC_LDS 0
#endif

static __device__ inline void async_ld_b128(const __bf16* g, __bf16* l) {
#if HAVE_ASYNC_LDS
    // signature (from compiler diagnostic): (int4 AS1* src, int4 AS3* dst, imm off, imm cpol)
    __builtin_amdgcn_global_load_async_to_lds_b128(
        (__attribute__((address_space(1))) i32x4*)(void*)g,
        (__attribute__((address_space(3))) i32x4*)(void*)l,
        /*offset=*/0, /*cpol=*/0);
#else
    *(bf16x8*)l = *(const bf16x8*)g;
#endif
}

template<int N>
static __device__ inline void wait_async() {
#if HAVE_ASYNC_LDS
#  if __has_builtin(__builtin_amdgcn_s_wait_asynccnt)
    __builtin_amdgcn_s_wait_asynccnt((unsigned short)N);
#  else
    asm volatile("s_wait_asynccnt %0" :: "n"(N));
#  endif
#endif
}

// ---------------------------------------------------------------- f32 -> bf16
__global__ void cvt_f32_bf16(const float* __restrict__ in, __bf16* __restrict__ out, int n) {
    int i = blockIdx.x * blockDim.x + threadIdx.x;
    int stride = gridDim.x * blockDim.x;
    for (; i < n; i += stride) out[i] = (__bf16)in[i];
}

// ---------------------------------------------------------------- tiled WMMA GEMM
// C[M,N] = A[M,K] @ W[N,K]^T + bias[N]   (A,W bf16 row-major; acc f32)
// Block tile 128x128, K-step 32, 256 threads = 8 waves (4 along M x 2 along N),
// wave tile 32x64 = 2x4 WMMA accumulators. Double-buffered LDS; tiles stream in
// via GLOBAL_LOAD_ASYNC_TO_LDS (ASYNCcnt) when available.
template<bool RELU, bool OUTF32, bool OUTBF>
__global__ __launch_bounds__(256)
void gemm_bf16(const __bf16* __restrict__ A, const __bf16* __restrict__ W,
               const float* __restrict__ bias,
               float* __restrict__ Cf, __bf16* __restrict__ Cb,
               int M, int N, int K)
{
    constexpr int LDT = 40;                   // padded LDS row stride (bf16)
    __shared__ __bf16 As[2][128 * LDT];
    __shared__ __bf16 Ws[2][128 * LDT];

    const int tid  = threadIdx.x;
    const int lane = tid & 31;
    const int wv   = tid >> 5;
    const int wm   = wv & 3;                  // 0..3  (M)
    const int wn   = wv >> 2;                 // 0..1  (N)
    const int l16  = lane & 15;
    const int lh   = lane >> 4;
    const int bm   = blockIdx.y * 128;
    const int bn   = blockIdx.x * 128;

    const int ldRow  = tid >> 1;              // 0..127
    const int ldHalf = (tid & 1) * 16;        // 0 / 16 (bf16 elems)

    f32x8 acc[2][4];
#pragma unroll
    for (int mt = 0; mt < 2; ++mt)
#pragma unroll
        for (int nt = 0; nt < 4; ++nt) acc[mt][nt] = zero8();

    // stage one 128x32 A tile + 128x32 W tile into LDS buffer `buf`
    auto stage = [&](int k0, int buf) {
        const __bf16* ga = A + (size_t)(bm + ldRow) * K + k0 + ldHalf;
        const __bf16* gw = W + (size_t)(bn + ldRow) * K + k0 + ldHalf;
        __bf16* la = &As[buf][ldRow * LDT + ldHalf];
        __bf16* lw = &Ws[buf][ldRow * LDT + ldHalf];
        async_ld_b128(ga,     la);        // 4 async-load instructions / wave / tile
        async_ld_b128(ga + 8, la + 8);
        async_ld_b128(gw,     lw);
        async_ld_b128(gw + 8, lw + 8);
    };

    const int nk = K >> 5;
    stage(0, 0);

    for (int ki = 0; ki < nk; ++ki) {
        const int buf = ki & 1;
        if (ki + 1 < nk) {
            stage((ki + 1) << 5, buf ^ 1);
            wait_async<4>();              // current tile done; next tile in flight
        } else {
            wait_async<0>();
        }
        __syncthreads();

        Frag16 fa[2], fb[4];
#pragma unroll
        for (int mt = 0; mt < 2; ++mt) {
            int row = wm * 32 + mt * 16 + l16;
            int kb  = lh * 16;                // contiguous k order, same as B: lanes16-31 hold K=16..31
            fa[mt].h[0] = *(const bf16x8*)&As[buf][row * LDT + kb];
            fa[mt].h[1] = *(const bf16x8*)&As[buf][row * LDT + kb + 8];
        }
#pragma unroll
        for (int nt = 0; nt < 4; ++nt) {
            int n  = wn * 64 + nt * 16 + l16;
            int kb = lh * 16;                 // contiguous k order, same as A: lanes16-31 hold K=16..31
            fb[nt].h[0] = *(const bf16x8*)&Ws[buf][n * LDT + kb];
            fb[nt].h[1] = *(const bf16x8*)&Ws[buf][n * LDT + kb + 8];
        }
#pragma unroll
        for (int mt = 0; mt < 2; ++mt)
#pragma unroll
            for (int nt = 0; nt < 4; ++nt)
                acc[mt][nt] = wmma_bf16(fa[mt].v, fb[nt].v, acc[mt][nt]);
        __syncthreads();                      // all waves done reading buf
    }

    // epilogue: bias (+relu), store f32 and/or bf16
#pragma unroll
    for (int mt = 0; mt < 2; ++mt)
#pragma unroll
        for (int nt = 0; nt < 4; ++nt) {
            int   col = bn + wn * 64 + nt * 16 + l16;
            float bv  = bias[col];
#pragma unroll
            for (int vg = 0; vg < 8; ++vg) {
                int   row = bm + wm * 32 + mt * 16 + vg + 8 * lh;
                float v   = acc[mt][nt][vg] + bv;
                if (RELU) v = fmaxf(v, 0.f);
                size_t idx = (size_t)row * N + col;
                if (OUTF32) Cf[idx] = v;
                if (OUTBF)  Cb[idx] = (__bf16)v;
            }
        }
}

// ---------------------------------------------------------------- fused attention
// One workgroup per (b, h, 64-row P tile); 4 waves, 16 query rows per wave.
// Whole K head (512x64) and V^T (64x512) live in LDS (320KB WGP LDS).
constexpr int KS_LD = 72;     // padded bf16 row stride for K   (512 rows)
constexpr int VT_LD = 520;    // padded bf16 row stride for V^T (64 rows)
constexpr int SS_LD = 516;    // padded f32 row stride for score tile (16 rows/wave)
constexpr size_t SM_KS = (size_t)512 * KS_LD * 2;   //  73728
constexpr size_t SM_VT = (size_t)64  * VT_LD * 2;   //  66560
constexpr size_t SM_SS = (size_t)4 * 16 * SS_LD * 4;// 132096
constexpr size_t SM_MK = 512;
constexpr size_t ATTN_SMEM = SM_KS + SM_VT + SM_SS + SM_MK;  // 272896 B < 320KB

__global__ __launch_bounds__(128)
void attn_fused(const __bf16* __restrict__ qbf, const __bf16* __restrict__ kvbf,
                const unsigned char* __restrict__ mask, const float* __restrict__ log_tau,
                float* __restrict__ wout, __bf16* __restrict__ attnbf)
{
    extern __shared__ char smem[];
    __bf16* Ks = (__bf16*)smem;
    __bf16* Vt = (__bf16*)(smem + SM_KS);
    float*  Ss = (float*)(smem + SM_KS + SM_VT);
    unsigned char* mk = (unsigned char*)(smem + SM_KS + SM_VT + SM_SS);

    const int tid  = threadIdx.x;
    const int lane = tid & 31;
    const int wv   = tid >> 5;
    const int l16  = lane & 15;
    const int lh   = lane >> 4;
    const int b    = blockIdx.z;
    const int h    = blockIdx.y;
    const int pt   = blockIdx.x;

    // ---- cooperative LDS fill: K via async global->LDS, V transposed via VGPRs
#pragma unroll 4
    for (int i = 0; i < 32; ++i) {
        int c = i * 128 + tid;                // 4096 chunks of 8 bf16
        int row = c >> 3, cc = (c & 7) * 8;
        async_ld_b128(&kvbf[(size_t)(b * Q + row) * (2 * D) + h * DH + cc],
                      &Ks[row * KS_LD + cc]);
    }
#pragma unroll 4
    for (int i = 0; i < 32; ++i) {
        int c = i * 128 + tid;
        int row = c >> 3, cc = (c & 7) * 8;
        bf16x8 v = *(const bf16x8*)&kvbf[(size_t)(b * Q + row) * (2 * D) + D + h * DH + cc];
#pragma unroll
        for (int j = 0; j < 8; ++j) Vt[(cc + j) * VT_LD + row] = v[j];
    }
#pragma unroll
    for (int j = 0; j < 4; ++j) mk[tid * 4 + j] = mask[b * Q + tid * 4 + j];
    wait_async<0>();
    __syncthreads();

    const int prow0 = pt * 64 + wv * 16;
    const float scale = __expf(-log_tau[h]) * 0.125f;   // /sqrt(64)/tau

    // ---- Q fragments straight from global (already bf16)
    Frag16 fq[2];
#pragma unroll
    for (int ks = 0; ks < 2; ++ks) {
        const __bf16* g = qbf + (size_t)(b * P + prow0 + l16) * D + h * DH + ks * 32 + lh * 16;
        fq[ks].h[0] = *(const bf16x8*)g;
        fq[ks].h[1] = *(const bf16x8*)(g + 8);
    }

    float* SsW = Ss + wv * 16 * SS_LD;   // this wave's 16x512 score tile

    // ---- scores: S = Q @ K^T, scale, mask, spill to LDS (f32)
    for (int nc = 0; nc < 32; ++nc) {
        f32x8 accS = zero8();
#pragma unroll
        for (int ks = 0; ks < 2; ++ks) {
            Frag16 fk;
            int n = nc * 16 + l16;
            fk.h[0] = *(const bf16x8*)&Ks[n * KS_LD + ks * 32 + lh * 16];
            fk.h[1] = *(const bf16x8*)&Ks[n * KS_LD + ks * 32 + lh * 16 + 8];
            accS = wmma_bf16(fq[ks].v, fk.v, accS);
        }
        int col = nc * 16 + l16;
        bool keep = mk[col] != 0;
#pragma unroll
        for (int vg = 0; vg < 8; ++vg) {
            float s = accS[vg] * scale;
            if (!keep) s = -__builtin_inff();
            SsW[(vg + 8 * lh) * SS_LD + col] = s;
        }
    }
    // DS ops of one wave are in-order: no barrier needed before reading back.

    // ---- exact softmax stats: 2 lanes per row (each scans 256 cols)
    float rmax, rinv;
    {
        const float* rp = SsW + l16 * SS_LD + lh * 256;
        float m = -__builtin_inff();
        for (int i = 0; i < 256; ++i) m = fmaxf(m, rp[i]);
        m = fmaxf(m, __shfl_xor(m, 16));
        float s = 0.f;
        for (int i = 0; i < 256; ++i) s += __expf(rp[i] - m);
        s += __shfl_xor(s, 16);
        rmax = m;
        rinv = 1.0f / s;
    }

    // ---- normalize: write f32 weights to d_out, compact bf16 in place in LDS
    float* outBase = wout + ((size_t)(b * H + h) * P + prow0) * Q;
    for (int it = 0; it < 256; ++it) {
        int row = it >> 4;                       // uniform across wave
        int col = (it & 15) * 32 + lane;
        float rm = __shfl(rmax, row);
        float ri = __shfl(rinv, row);
        float s  = SsW[row * SS_LD + col];
        float w  = __expf(s - rm) * ri;
        outBase[(size_t)row * Q + col] = w;
        ((__bf16*)(SsW + row * SS_LD))[col] = (__bf16)w;  // clobbers only consumed f32
    }

    // ---- attn = Pw(16x512) @ V(512x64)
    f32x8 accO[4];
#pragma unroll
    for (int nt = 0; nt < 4; ++nt) accO[nt] = zero8();
    for (int ks = 0; ks < 16; ++ks) {
        Frag16 fp;
        const __bf16* abase = (const __bf16*)(SsW + l16 * SS_LD);
        int kb = ks * 32 + lh * 16;
        fp.h[0] = *(const bf16x8*)(abase + kb);
        fp.h[1] = *(const bf16x8*)(abase + kb + 8);
#pragma unroll
        for (int nt = 0; nt < 4; ++nt) {
            Frag16 fv;
            int n = nt * 16 + l16;
            fv.h[0] = *(const bf16x8*)&Vt[n * VT_LD + ks * 32 + lh * 16];
            fv.h[1] = *(const bf16x8*)&Vt[n * VT_LD + ks * 32 + lh * 16 + 8];
            accO[nt] = wmma_bf16(fp.v, fv.v, accO[nt]);
        }
    }
    // store attn (already in transposed-back [b,p,h*64+dh] layout), bf16
#pragma unroll
    for (int nt = 0; nt < 4; ++nt)
#pragma unroll
        for (int vg = 0; vg < 8; ++vg) {
            int prow = prow0 + vg + 8 * lh;
            int col  = h * DH + nt * 16 + l16;
            attnbf[(size_t)(b * P + prow) * D + col] = (__bf16)accO[nt][vg];
        }
}

// ---------------------------------------------------------------- layernorms
__device__ inline float block_reduce(float v, float* red, int tid) {
    red[tid] = v; __syncthreads();
    for (int o = 128; o > 0; o >>= 1) {
        if (tid < o) red[tid] += red[tid + o];
        __syncthreads();
    }
    float r = red[0]; __syncthreads();
    return r;
}

__global__ __launch_bounds__(256)
void ln1_kernel(const float* __restrict__ x0, const float* __restrict__ up,
                const float* __restrict__ g, const float* __restrict__ bb,
                float* __restrict__ y32, __bf16* __restrict__ ybf)
{
    __shared__ float red[256];
    const int r = blockIdx.x, tid = threadIdx.x;
    const float4 va = ((const float4*)(x0 + (size_t)r * D))[tid];
    const float4 vb = ((const float4*)(up + (size_t)r * D))[tid];
    float v[4] = {va.x + vb.x, va.y + vb.y, va.z + vb.z, va.w + vb.w};
    float s = v[0] + v[1] + v[2] + v[3];
    float mu = block_reduce(s, red, tid) * (1.0f / D);
    float q = 0.f;
#pragma unroll
    for (int j = 0; j < 4; ++j) { float d = v[j] - mu; q += d * d; }
    float var = block_reduce(q, red, tid) * (1.0f / D);
    float rstd = rsqrtf(var + 1e-5f);
#pragma unroll
    for (int j = 0; j < 4; ++j) {
        int i = tid * 4 + j;
        float y = (v[j] - mu) * rstd * g[i] + bb[i];
        y32[(size_t)r * D + i] = y;
        ybf[(size_t)r * D + i] = (__bf16)y;
    }
}

__global__ __launch_bounds__(256)
void ln2_cls_kernel(const float* __restrict__ x1, const float* __restrict__ ff,
                    const float* __restrict__ g, const float* __restrict__ bb,
                    const float* __restrict__ cw, const float* __restrict__ cb,
                    float* __restrict__ xout, float* __restrict__ logits,
                    float* __restrict__ probs)
{
    __shared__ float red[256];
    const int r = blockIdx.x, tid = threadIdx.x;
    const float4 va = ((const float4*)(x1 + (size_t)r * D))[tid];
    const float4 vb = ((const float4*)(ff + (size_t)r * D))[tid];
    float v[4] = {va.x + vb.x, va.y + vb.y, va.z + vb.z, va.w + vb.w};
    float s = v[0] + v[1] + v[2] + v[3];
    float mu = block_reduce(s, red, tid) * (1.0f / D);
    float q = 0.f;
#pragma unroll
    for (int j = 0; j < 4; ++j) { float d = v[j] - mu; q += d * d; }
    float var = block_reduce(q, red, tid) * (1.0f / D);
    float rstd = rsqrtf(var + 1e-5f);
    float p = 0.f;
#pragma unroll
    for (int j = 0; j < 4; ++j) {
        int i = tid * 4 + j;
        float y = (v[j] - mu) * rstd * g[i] + bb[i];
        xout[(size_t)r * D + i] = y;
        p += y * cw[i];
    }
    float lg = block_reduce(p, red, tid);
    if (tid == 0) {
        lg += cb[0];
        logits[r] = lg;
        probs[r]  = 1.0f / (1.0f + __expf(-lg));
    }
}

// ---------------------------------------------------------------- launcher
// workspace layout (bytes)
constexpr size_t WS_IMG_BF  = 0;                       // B*P*D bf16
constexpr size_t WS_TXT_BF  = 33554432;                // B*Q*D bf16
constexpr size_t WS_WIN_BF  = 50331648;                // 3D*D bf16
constexpr size_t WS_WOUT_BF = 56623104;                // D*D bf16
constexpr size_t WS_WF1_BF  = 58720256;                // 2D*D bf16
constexpr size_t WS_WF2_BF  = 62914560;                // D*2D bf16
constexpr size_t WS_Q_BF    = 67108864;                // B*P*D bf16
constexpr size_t WS_KV_BF   = 100663296;               // B*Q*2D bf16
constexpr size_t WS_ATTN_BF = 134217728;               // B*P*D bf16
constexpr size_t WS_GEMMF32 = 167772160;               // B*P*D f32 (img_up, then ff)
constexpr size_t WS_X1_F32  = 234881024;               // B*P*D f32
constexpr size_t WS_X1_BF   = 301989888;               // B*P*D bf16
constexpr size_t WS_H1_BF   = 335544320;               // B*P*2D bf16

// d_out layout (floats)
constexpr size_t OUT_X  = 0;
constexpr size_t OUT_W  = (size_t)B * P * D;                 //  16777216
constexpr size_t OUT_LG = OUT_W + (size_t)B * H * P * Q;     // 150994944
constexpr size_t OUT_PR = OUT_LG + (size_t)B * P;            // 151011328

extern "C" void kernel_launch(void* const* d_in, const int* in_sizes, int n_in,
                              void* d_out, int out_size, void* d_ws, size_t ws_size,
                              hipStream_t stream) {
    (void)in_sizes; (void)n_in; (void)out_size; (void)ws_size;

    const float* img   = (const float*)d_in[0];
    const float* txt   = (const float*)d_in[1];
    const unsigned char* msk = (const unsigned char*)d_in[2];
    const float* w_in  = (const float*)d_in[3];
    const float* b_in  = (const float*)d_in[4];
    const float* w_out = (const float*)d_in[5];
    const float* b_out = (const float*)d_in[6];
    const float* ltau  = (const float*)d_in[7];
    const float* n1g   = (const float*)d_in[8];
    const float* n1b   = (const float*)d_in[9];
    const float* wf1   = (const float*)d_in[10];
    const float* bf1   = (const float*)d_in[11];
    const float* wf2   = (const float*)d_in[12];
    const float* bf2   = (const float*)d_in[13];
    const float* n2g   = (const float*)d_in[14];
    const float* n2b   = (const float*)d_in[15];
    const float* clsw  = (const float*)d_in[16];
    const float* clsb  = (const float*)d_in[17];

    char* ws = (char*)d_ws;
    __bf16* img_bf  = (__bf16*)(ws + WS_IMG_BF);
    __bf16* txt_bf  = (__bf16*)(ws + WS_TXT_BF);
    __bf16* win_bf  = (__bf16*)(ws + WS_WIN_BF);
    __bf16* wout_bf = (__bf16*)(ws + WS_WOUT_BF);
    __bf16* wf1_bf  = (__bf16*)(ws + WS_WF1_BF);
    __bf16* wf2_bf  = (__bf16*)(ws + WS_WF2_BF);
    __bf16* q_bf    = (__bf16*)(ws + WS_Q_BF);
    __bf16* kv_bf   = (__bf16*)(ws + WS_KV_BF);
    __bf16* attn_bf = (__bf16*)(ws + WS_ATTN_BF);
    float*  gemmf   = (float*)(ws + WS_GEMMF32);
    float*  x1f     = (float*)(ws + WS_X1_F32);
    __bf16* x1bf    = (__bf16*)(ws + WS_X1_BF);
    __bf16* h1_bf   = (__bf16*)(ws + WS_H1_BF);

    float* out = (float*)d_out;

    auto cvt = [&](const float* src, __bf16* dst, int n) {
        int blocks = (n + 1023) / 1024;
        if (blocks > 65535) blocks = 65535;
        cvt_f32_bf16<<<blocks, 1024, 0, stream>>>(src, dst, n);
    };
    cvt(img,   img_bf,  B * P * D);
    cvt(txt,   txt_bf,  B * Q * D);
    cvt(w_in,  win_bf,  3 * D * D);
    cvt(w_out, wout_bf, D * D);
    cvt(wf1,   wf1_bf,  2 * D * D);
    cvt(wf2,   wf2_bf,  D * 2 * D);

    // q = img @ Wq^T + bq        (M=16384, N=1024, K=1024) -> bf16
    gemm_bf16<false, false, true><<<dim3(D / 128, (B * P) / 128), 256, 0, stream>>>(
        img_bf, win_bf, b_in, nullptr, q_bf, B * P, D, D);
    // kv = txt @ Wkv^T + bkv     (M=8192, N=2048, K=1024) -> bf16
    gemm_bf16<false, false, true><<<dim3((2 * D) / 128, (B * Q) / 128), 256, 0, stream>>>(
        txt_bf, win_bf + (size_t)D * D, b_in + D, nullptr, kv_bf, B * Q, 2 * D, D);

    // fused attention (scores + softmax + weights-out + P@V)
    (void)hipFuncSetAttribute((const void*)attn_fused,
                              hipFuncAttributeMaxDynamicSharedMemorySize,
                              (int)ATTN_SMEM);
    attn_fused<<<dim3(P / 64, H, B), 128, ATTN_SMEM, stream>>>(
        q_bf, kv_bf, msk, ltau, out + OUT_W, attn_bf);

    // img_up = attn @ out_w^T + out_b -> f32
    gemm_bf16<false, true, false><<<dim3(D / 128, (B * P) / 128), 256, 0, stream>>>(
        attn_bf, wout_bf, b_out, gemmf, nullptr, B * P, D, D);

    // x1 = LN(img + img_up)
    ln1_kernel<<<B * P, 256, 0, stream>>>(img, gemmf, n1g, n1b, x1f, x1bf);

    // h1 = relu(x1 @ w1^T + b1) -> bf16   (N=2048)
    gemm_bf16<true, false, true><<<dim3((2 * D) / 128, (B * P) / 128), 256, 0, stream>>>(
        x1bf, wf1_bf, bf1, nullptr, h1_bf, B * P, 2 * D, D);
    // ff = h1 @ w2^T + b2 -> f32          (K=2048)
    gemm_bf16<false, true, false><<<dim3(D / 128, (B * P) / 128), 256, 0, stream>>>(
        h1_bf, wf2_bf, bf2, gemmf, nullptr, B * P, D, 2 * D);

    // x = LN(x1 + ff); logits = x @ cls_w^T + cls_b; sigmoid
    ln2_cls_kernel<<<B * P, 256, 0, stream>>>(
        x1f, gemmf, n2g, n2b, clsw, clsb, out + OUT_X, out + OUT_LG, out + OUT_PR);
}
